// TransformerBlock_32469952758515
// MI455X (gfx1250) — compile-verified, matrix operand order checked by reading, not run on hardware
//
#include <hip/hip_runtime.h>
#include <hip/hip_bf16.h>

// ---------------- model constants (from reference) ----------------
namespace cfg {
constexpr int H    = 2048;
constexpr int NH   = 16;
constexpr int NOPE = 128;
constexpr int ROPE = 64;
constexpr int HALF = ROPE / 2;     // 32
constexpr int HD   = NOPE + ROPE;  // 192
constexpr int VD   = 128;
constexpr int QL   = 1536;
constexpr int KVL  = 512;
constexpr int E    = 8;
constexpr int MI   = 512;
constexpr int SHI  = 2 * MI;       // 1024
constexpr int T    = 2048;         // B*S, B==1
constexpr int S    = 2048;
}

typedef __attribute__((ext_vector_type(16))) _Float16 v16h;
typedef __attribute__((ext_vector_type(8)))  _Float16 v8h;
typedef __attribute__((ext_vector_type(4)))  _Float16 v4h;
typedef __attribute__((ext_vector_type(8)))  float    v8f;

// =====================================================================
// Generic WMMA GEMM:  C[M,N] = A[M,K] * B[N,K]^T   (fp32 in/out, f16 MAC)
//   A element (m,k) at  A + m*a_rs + k            (k-contiguous)
//   B element (n,k) at  B + n*b_ns + k*b_ks       (b_ks!=1 gives V^T views)
//   C element (m,n) at  C + m*c_rs + n
// Block tile 128x128, BK=32; 8 waves (4x2), each wave owns 32x64:
// 2 A-frags x 4 B-frags -> 8 v_wmma_f32_16x16x32_f16 per K step.
// Fast path (interior tiles, K%32==0): DOUBLE-BUFFERED LDS pipeline —
// next-tile global_load_b128 clause issues before the current tile's
// WMMAs, packed f16 converts + ds stores go to the alternate buffer,
// one barrier per K step. Edge path (the N=8 gate GEMM) uses CLAMPED
// unconditional loads + cndmask zeroing — no exec-mask serialization.
// =====================================================================
#define GBM 128
#define GBN 128
#define GBK 32
#define LDA_SH 40   // 40 halves = 80B row stride -> 16B aligned chunks
#define LDB_SH 40

__global__ __launch_bounds__(256)
void gemm_f16_wmma(const float* __restrict__ A, long a_rs,
                   const float* __restrict__ B, long b_ns, long b_ks,
                   float* __restrict__ C, long c_rs,
                   int M, int N, int K)
{
    __shared__ _Float16 sA[2][GBM * LDA_SH] __attribute__((aligned(16)));
    __shared__ _Float16 sB[2][GBN * LDB_SH] __attribute__((aligned(16)));

    const int tid  = threadIdx.x;
    const int lane = tid & 31;
    const int wave = tid >> 5;     // 0..7
    const int wm   = wave & 3;     // 4 waves along M (32 rows each)
    const int wn   = wave >> 2;    // 2 waves along N (64 cols each)
    const int l15  = lane & 15;
    const int kg   = (lane < 16) ? 0 : 8;   // ISA 16-bit A/B fragment K group

    const int bm = blockIdx.y * GBM;
    const int bn = blockIdx.x * GBN;

    // staging coordinates: thread covers col quad colA, rows rowA + p*32
    const int rowA = tid >> 3;         // 0..31
    const int colA = (tid & 7) * 4;    // 0,4,...,28

    const bool interior = (bm + GBM <= M) && (bn + GBN <= N);

    v8f acc[2][4];
    #pragma unroll
    for (int i = 0; i < 2; ++i)
        #pragma unroll
        for (int j = 0; j < 4; ++j)
            acc[i][j] = (v8f)(0.0f);

    if (interior && (K % GBK == 0)) {
        // =================== fast double-buffered pipeline ===================
        float4 va[4], vb[4];

        // ---- stage tile 0 into registers
        #pragma unroll
        for (int p = 0; p < 4; ++p)
            va[p] = *(const float4*)(A + (long)(bm + rowA + p * 32) * a_rs + colA);
        if (b_ks == 1) {
            #pragma unroll
            for (int p = 0; p < 4; ++p)
                vb[p] = *(const float4*)(B + (long)(bn + rowA + p * 32) * b_ns + colA);
        } else {
            #pragma unroll
            for (int p = 0; p < 4; ++p) {
                const float* bp = B + (long)(bn + rowA + p * 32) * b_ns
                                    + (long)colA * b_ks;
                vb[p].x = bp[0];
                vb[p].y = bp[b_ks];
                vb[p].z = bp[2 * b_ks];
                vb[p].w = bp[3 * b_ks];
            }
        }
        // ---- convert + store tile 0 into buffer 0
        #pragma unroll
        for (int p = 0; p < 4; ++p) {
            const int r = rowA + p * 32;
            v4h ha = { (_Float16)va[p].x, (_Float16)va[p].y,
                       (_Float16)va[p].z, (_Float16)va[p].w };
            v4h hb = { (_Float16)vb[p].x, (_Float16)vb[p].y,
                       (_Float16)vb[p].z, (_Float16)vb[p].w };
            *(v4h*)&sA[0][r * LDA_SH + colA] = ha;
            *(v4h*)&sB[0][r * LDB_SH + colA] = hb;
        }
        __syncthreads();

        int buf = 0;
        for (int k0 = 0; k0 < K; k0 += GBK) {
            const bool has_next = (k0 + GBK) < K;

            // ---- issue next-tile global loads (overlap with WMMAs below)
            if (has_next) {
                const int kn = k0 + GBK;
                #pragma unroll
                for (int p = 0; p < 4; ++p)
                    va[p] = *(const float4*)(A + (long)(bm + rowA + p * 32) * a_rs
                                               + kn + colA);
                if (b_ks == 1) {
                    #pragma unroll
                    for (int p = 0; p < 4; ++p)
                        vb[p] = *(const float4*)(B + (long)(bn + rowA + p * 32) * b_ns
                                                   + kn + colA);
                } else {
                    #pragma unroll
                    for (int p = 0; p < 4; ++p) {
                        const float* bp = B + (long)(bn + rowA + p * 32) * b_ns
                                            + (long)(kn + colA) * b_ks;
                        vb[p].x = bp[0];
                        vb[p].y = bp[b_ks];
                        vb[p].z = bp[2 * b_ks];
                        vb[p].w = bp[3 * b_ks];
                    }
                }
                if (kn + GBK < K)   // warm L2 two tiles ahead
                    __builtin_prefetch(A + (long)(bm + rowA) * a_rs + kn + GBK + colA,
                                       0, 1);
            }

            // ---- compute current buffer: 12 ds_load_b128 -> 8 WMMAs
            {
                const _Float16* cA = sA[buf];
                const _Float16* cB = sB[buf];
                v16h af[2], bf[4];
                #pragma unroll
                for (int i = 0; i < 2; ++i) {
                    const int ar = wm * 32 + i * 16 + l15;
                    v8h lo = *(const v8h*)&cA[ar * LDA_SH + kg];
                    v8h hi = *(const v8h*)&cA[ar * LDA_SH + kg + 16];
                    af[i] = __builtin_shufflevector(lo, hi,
                                0,1,2,3,4,5,6,7,8,9,10,11,12,13,14,15);
                }
                #pragma unroll
                for (int j = 0; j < 4; ++j) {
                    const int br = wn * 64 + j * 16 + l15;
                    v8h lo = *(const v8h*)&cB[br * LDB_SH + kg];
                    v8h hi = *(const v8h*)&cB[br * LDB_SH + kg + 16];
                    bf[j] = __builtin_shufflevector(lo, hi,
                                0,1,2,3,4,5,6,7,8,9,10,11,12,13,14,15);
                }
                #pragma unroll
                for (int i = 0; i < 2; ++i)
                    #pragma unroll
                    for (int j = 0; j < 4; ++j)
                        acc[i][j] = __builtin_amdgcn_wmma_f32_16x16x32_f16(
                            false, af[i], false, bf[j], (short)0, acc[i][j],
                            false, false);
            }

            // ---- convert + store next tile into alternate buffer
            if (has_next) {
                const int nb = buf ^ 1;
                #pragma unroll
                for (int p = 0; p < 4; ++p) {
                    const int r = rowA + p * 32;
                    v4h ha = { (_Float16)va[p].x, (_Float16)va[p].y,
                               (_Float16)va[p].z, (_Float16)va[p].w };
                    v4h hb = { (_Float16)vb[p].x, (_Float16)vb[p].y,
                               (_Float16)vb[p].z, (_Float16)vb[p].w };
                    *(v4h*)&sA[nb][r * LDA_SH + colA] = ha;
                    *(v4h*)&sB[nb][r * LDB_SH + colA] = hb;
                }
            }
            __syncthreads();
            buf ^= 1;
        }
    } else {
        // ============ edge-tile loop: clamped unconditional loads ============
        // Clamp indices to legal addresses, load unconditionally, zero via
        // select (v_cndmask) — no exec-mask branches, loads can cluster.
        for (int k0 = 0; k0 < K; k0 += GBK) {
            #pragma unroll
            for (int p = 0; p < 4; ++p) {
                const int r   = rowA + p * 32;
                const int gr  = bm + r;
                const int grc = gr < M ? gr : M - 1;
                _Float16* dst = &sA[0][r * LDA_SH + colA];
                float fv[4];
                #pragma unroll
                for (int j = 0; j < 4; ++j) {
                    const int gk  = k0 + colA + j;
                    const int gkc = gk < K ? gk : K - 1;
                    fv[j] = A[(long)grc * a_rs + gkc];
                }
                v4h hv;
                #pragma unroll
                for (int j = 0; j < 4; ++j) {
                    const int gk = k0 + colA + j;
                    hv[j] = (_Float16)((gr < M && gk < K) ? fv[j] : 0.0f);
                }
                *(v4h*)dst = hv;
            }
            #pragma unroll
            for (int p = 0; p < 4; ++p) {
                const int r   = rowA + p * 32;
                const int gn  = bn + r;
                const int gnc = gn < N ? gn : N - 1;
                _Float16* dst = &sB[0][r * LDB_SH + colA];
                float fv[4];
                #pragma unroll
                for (int j = 0; j < 4; ++j) {
                    const int gk  = k0 + colA + j;
                    const int gkc = gk < K ? gk : K - 1;
                    fv[j] = B[(long)gnc * b_ns + (long)gkc * b_ks];
                }
                v4h hv;
                #pragma unroll
                for (int j = 0; j < 4; ++j) {
                    const int gk = k0 + colA + j;
                    hv[j] = (_Float16)((gn < N && gk < K) ? fv[j] : 0.0f);
                }
                *(v4h*)dst = hv;
            }
            __syncthreads();

            v16h af[2], bf[4];
            #pragma unroll
            for (int i = 0; i < 2; ++i) {
                const int ar = wm * 32 + i * 16 + l15;
                v8h lo = *(const v8h*)&sA[0][ar * LDA_SH + kg];
                v8h hi = *(const v8h*)&sA[0][ar * LDA_SH + kg + 16];
                af[i] = __builtin_shufflevector(lo, hi,
                            0,1,2,3,4,5,6,7,8,9,10,11,12,13,14,15);
            }
            #pragma unroll
            for (int j = 0; j < 4; ++j) {
                const int br = wn * 64 + j * 16 + l15;
                v8h lo = *(const v8h*)&sB[0][br * LDB_SH + kg];
                v8h hi = *(const v8h*)&sB[0][br * LDB_SH + kg + 16];
                bf[j] = __builtin_shufflevector(lo, hi,
                            0,1,2,3,4,5,6,7,8,9,10,11,12,13,14,15);
            }
            #pragma unroll
            for (int i = 0; i < 2; ++i)
                #pragma unroll
                for (int j = 0; j < 4; ++j)
                    acc[i][j] = __builtin_amdgcn_wmma_f32_16x16x32_f16(
                        false, af[i], false, bf[j], (short)0, acc[i][j],
                        false, false);
            __syncthreads();
        }
    }

    // ---- epilogue per ISA C/D layout: VGPR r -> M row r (lanes<16) / r+8
    #pragma unroll
    for (int i = 0; i < 2; ++i) {
        const int row0 = bm + wm * 32 + i * 16 + ((lane < 16) ? 0 : 8);
        #pragma unroll
        for (int j = 0; j < 4; ++j) {
            const int col = bn + wn * 64 + j * 16 + l15;
            #pragma unroll
            for (int r = 0; r < 8; ++r) {
                const int row = row0 + r;
                if (row < M && col < N)
                    C[(long)row * c_rs + col] = acc[i][j][r];
            }
        }
    }
}

// =====================================================================
// RMSNorm (optionally fused residual add): out = rms(x[+skip]) * w
// one block per row; strided so it handles the ckv 512-of-576 slice
// =====================================================================
__global__ __launch_bounds__(256)
void rmsnorm_kernel(const float* __restrict__ x, const float* __restrict__ skip,
                    const float* __restrict__ w, float* __restrict__ x1out,
                    float* __restrict__ out, int width, long in_rs, long out_rs)
{
    const long row = blockIdx.x;
    const float* xi = x + row * in_rs;
    const float* sk = skip ? skip + row * in_rs : nullptr;
    __shared__ float red[256];
    float ss = 0.0f;
    for (int i = threadIdx.x; i < width; i += 256) {
        float v = xi[i] + (sk ? sk[i] : 0.0f);
        if (x1out) x1out[row * in_rs + i] = v;
        ss += v * v;
    }
    red[threadIdx.x] = ss;
    __syncthreads();
    for (int s = 128; s > 0; s >>= 1) {
        if (threadIdx.x < s) red[threadIdx.x] += red[threadIdx.x + s];
        __syncthreads();
    }
    const float inv = rsqrtf(red[0] / (float)width + 1e-6f);
    for (int i = threadIdx.x; i < width; i += 256) {
        float v = xi[i] + (sk ? sk[i] : 0.0f);
        out[row * out_rs + i] = v * inv * w[i];
    }
}

// interleaved RoPE, in place on x[t, h, off + 2i / 2i+1]
__global__ void rope_kernel(float* __restrict__ x,
                            const float* __restrict__ sn,
                            const float* __restrict__ cs,
                            int total, int nh, int hd, int off)
{
    int idx = blockIdx.x * blockDim.x + threadIdx.x;
    if (idx >= total) return;
    const int i = idx & 31;
    const int h = (idx >> 5) % nh;
    const int t = idx / (32 * nh);
    const float s = sn[t * cfg::HALF + i];
    const float c = cs[t * cfg::HALF + i];
    float* p = x + (long)t * nh * hd + (long)h * hd + off + 2 * i;
    const float a = p[0], b = p[1];
    p[0] = a * c - b * s;
    p[1] = a * s + b * c;
}

// kf[t,h,:] = concat(k_nope[t,h,0:128], k_pe[t,0:64])  (k_pe lives in kv[:,512:])
__global__ void build_kf_kernel(const float* __restrict__ k_nope,
                                const float* __restrict__ kv,
                                float* __restrict__ kf, int total)
{
    int idx = blockIdx.x * blockDim.x + threadIdx.x;
    if (idx >= total) return;
    const int d = idx % cfg::HD;
    const int h = (idx / cfg::HD) % cfg::NH;
    const int t = idx / (cfg::HD * cfg::NH);
    kf[idx] = (d < cfg::NOPE)
            ? k_nope[(long)t * cfg::NH * cfg::NOPE + h * cfg::NOPE + d]
            : kv[(long)t * (cfg::KVL + cfg::ROPE) + cfg::KVL + (d - cfg::NOPE)];
}

// causal softmax over one score row (query q attends to k<=q); scale folded in
__global__ __launch_bounds__(256)
void softmax_causal_kernel(float* __restrict__ srow, int Sn, float scale)
{
    const int q = blockIdx.x;
    float* row = srow + (long)q * Sn;
    const int n = q + 1;
    __shared__ float red[256];
    float mx = -1e30f;
    for (int i = threadIdx.x; i < n; i += 256)
        mx = fmaxf(mx, row[i] * scale);
    red[threadIdx.x] = mx;
    __syncthreads();
    for (int s = 128; s > 0; s >>= 1) {
        if (threadIdx.x < s)
            red[threadIdx.x] = fmaxf(red[threadIdx.x], red[threadIdx.x + s]);
        __syncthreads();
    }
    mx = red[0];
    __syncthreads();
    float sum = 0.0f;
    for (int i = threadIdx.x; i < n; i += 256)
        sum += __expf(row[i] * scale - mx);
    red[threadIdx.x] = sum;
    __syncthreads();
    for (int s = 128; s > 0; s >>= 1) {
        if (threadIdx.x < s) red[threadIdx.x] += red[threadIdx.x + s];
        __syncthreads();
    }
    const float inv = 1.0f / red[0];
    for (int i = threadIdx.x; i < Sn; i += 256)
        row[i] = (i < n) ? __expf(row[i] * scale - mx) * inv : 0.0f;
}

// gate: softmax over 8 logits, top-2, renormalize -> dense weights we[t,e]
__global__ void gate_topk_kernel(const float* __restrict__ logits,
                                 float* __restrict__ we, int Tn)
{
    int t = blockIdx.x * blockDim.x + threadIdx.x;
    if (t >= Tn) return;
    float p[cfg::E];
    float mx = -1e30f;
    for (int e = 0; e < cfg::E; ++e) {
        p[e] = logits[t * cfg::E + e];
        mx = fmaxf(mx, p[e]);
    }
    float sum = 0.0f;
    for (int e = 0; e < cfg::E; ++e) { p[e] = __expf(p[e] - mx); sum += p[e]; }
    for (int e = 0; e < cfg::E; ++e) p[e] /= sum;
    int i0 = 0;
    for (int e = 1; e < cfg::E; ++e) if (p[e] > p[i0]) i0 = e;
    int i1 = (i0 == 0) ? 1 : 0;
    for (int e = 0; e < cfg::E; ++e)
        if (e != i0 && p[e] > p[i1]) i1 = e;
    const float s2 = p[i0] + p[i1];
    for (int e = 0; e < cfg::E; ++e) we[t * cfg::E + e] = 0.0f;
    we[t * cfg::E + i0] = p[i0] / s2;   // * MOE_SCALE (=1.0)
    we[t * cfg::E + i1] = p[i1] / s2;
}

// g = silu(h[:, :inner]) * h[:, inner:]
__global__ void silu_mul_kernel(const float* __restrict__ h,
                                float* __restrict__ g, int total, int inner)
{
    int idx = blockIdx.x * blockDim.x + threadIdx.x;
    if (idx >= total) return;
    const int t = idx / inner;
    const int i = idx % inner;
    const float a = h[(long)t * (2 * inner) + i];
    const float b = h[(long)t * (2 * inner) + inner + i];
    g[idx] = (a / (1.0f + __expf(-a))) * b;
}

// routed[t,:] += we[t,e] * hd[t,:]
__global__ void scale_add_kernel(float* __restrict__ routed,
                                 const float* __restrict__ hd,
                                 const float* __restrict__ we,
                                 int e, int total, int Hn)
{
    int idx = blockIdx.x * blockDim.x + threadIdx.x;
    if (idx >= total) return;
    const int t = idx / Hn;
    routed[idx] += we[t * cfg::E + e] * hd[idx];
}

__global__ void add_kernel(const float* __restrict__ a,
                           const float* __restrict__ b,
                           float* __restrict__ o, int total)
{
    int idx = blockIdx.x * blockDim.x + threadIdx.x;
    if (idx < total) o[idx] = a[idx] + b[idx];
}

__global__ void zero_kernel(float* __restrict__ p, int total)
{
    int idx = blockIdx.x * blockDim.x + threadIdx.x;
    if (idx < total) p[idx] = 0.0f;
}

// =====================================================================
// host orchestration
// =====================================================================
extern "C" void kernel_launch(void* const* d_in, const int* in_sizes, int n_in,
                              void* d_out, int out_size, void* d_ws, size_t ws_size,
                              hipStream_t stream)
{
    using namespace cfg;
    const float* x        = (const float*)d_in[0];
    const float* skip     = (const float*)d_in[1];
    const float* rsin     = (const float*)d_in[2];
    const float* rcos     = (const float*)d_in[3];
    const float* ln1_w    = (const float*)d_in[4];
    const float* ln2_w    = (const float*)d_in[5];
    const float* w_qa     = (const float*)d_in[6];   // [QL, H]
    const float* qa_nw    = (const float*)d_in[7];
    const float* w_qb     = (const float*)d_in[8];   // [NH*HD, QL]
    const float* w_kva    = (const float*)d_in[9];   // [KVL+ROPE, H]
    const float* kva_nw   = (const float*)d_in[10];
    const float* w_kb     = (const float*)d_in[11];  // [NH*NOPE, KVL]
    const float* w_vb     = (const float*)d_in[12];  // [NH*VD, KVL]
    const float* w_o      = (const float*)d_in[13];  // [H, NH*VD]
    const float* gate_w   = (const float*)d_in[14];  // [E, H]
    const float* moe_up   = (const float*)d_in[15];  // [E, 2*MI, H]
    const float* moe_dn   = (const float*)d_in[16];  // [E, H, MI]
    const float* sh_up    = (const float*)d_in[17];  // [2*SHI, H]
    const float* sh_dn    = (const float*)d_in[18];  // [H, SHI]

    float* out_x2  = (float*)d_out;            // [T, H]
    float* out_ffn = (float*)d_out + (long)T * H;

    // workspace carve-up
    float* f = (float*)d_ws;
    size_t o = 0;
    auto alloc = [&](size_t n) { float* p = f + o; o += n; return p; };
    float* x1     = alloc((size_t)T * H);
    float* n1     = alloc((size_t)T * H);
    float* qaraw  = alloc((size_t)T * QL);
    float* qa     = alloc((size_t)T * QL);
    float* q      = alloc((size_t)T * NH * HD);     // qf after in-place rope
    float* kv     = alloc((size_t)T * (KVL + ROPE));
    float* ckv    = alloc((size_t)T * KVL);
    float* knope  = alloc((size_t)T * NH * NOPE);
    float* v      = alloc((size_t)T * NH * VD);
    float* kf     = alloc((size_t)T * NH * HD);
    float* scores = alloc((size_t)S * S);           // reused per head
    float* obuf   = alloc((size_t)T * NH * VD);
    float* attn   = alloc((size_t)T * H);
    float* n2     = alloc((size_t)T * H);
    float* logits = alloc((size_t)T * E);
    float* we     = alloc((size_t)T * E);
    float* hup    = alloc((size_t)T * 2 * SHI);     // covers MoE (1024) & shared (2048)
    float* hg     = alloc((size_t)T * SHI);
    float* hdown  = alloc((size_t)T * H);
    float* routed = alloc((size_t)T * H);
    (void)ws_size; (void)n_in; (void)in_sizes; (void)out_size;

    auto gemm = [&](const float* A, long a_rs, const float* B, long b_ns,
                    long b_ks, float* C, long c_rs, int M, int N, int K) {
        dim3 g((N + GBN - 1) / GBN, (M + GBM - 1) / GBM);
        gemm_f16_wmma<<<g, 256, 0, stream>>>(A, a_rs, B, b_ns, b_ks, C, c_rs, M, N, K);
    };
    auto grid1 = [](int n) { return dim3((n + 255) / 256); };

    // ---------- attention ----------
    rmsnorm_kernel<<<T, 256, 0, stream>>>(x, skip, ln1_w, x1, n1, H, H, H);

    gemm(n1, H, w_qa, H, 1, qaraw, QL, T, QL, H);
    rmsnorm_kernel<<<T, 256, 0, stream>>>(qaraw, nullptr, qa_nw, nullptr, qa, QL, QL, QL);
    gemm(qa, QL, w_qb, QL, 1, q, NH * HD, T, NH * HD, QL);

    gemm(n1, H, w_kva, H, 1, kv, KVL + ROPE, T, KVL + ROPE, H);
    rmsnorm_kernel<<<T, 256, 0, stream>>>(kv, nullptr, kva_nw, nullptr, ckv,
                                          KVL, KVL + ROPE, KVL);

    // rope: q_pe (16 heads) and k_pe (the 64 tail floats of kv rows)
    rope_kernel<<<grid1(T * NH * HALF), 256, 0, stream>>>(q, rsin, rcos,
                                                          T * NH * HALF, NH, HD, NOPE);
    rope_kernel<<<grid1(T * HALF), 256, 0, stream>>>(kv, rsin, rcos,
                                                     T * HALF, 1, KVL + ROPE, KVL);

    gemm(ckv, KVL, w_kb, KVL, 1, knope, NH * NOPE, T, NH * NOPE, KVL);
    gemm(ckv, KVL, w_vb, KVL, 1, v, NH * VD, T, NH * VD, KVL);

    build_kf_kernel<<<grid1(T * NH * HD), 256, 0, stream>>>(knope, kv, kf, T * NH * HD);

    const float scale = rsqrtf((float)HD);
    for (int h = 0; h < NH; ++h) {
        // scores = q_h @ kf_h^T     [S, S], K = 192
        gemm(q + h * HD, NH * HD, kf + h * HD, NH * HD, 1,
             scores, S, S, S, HD);
        softmax_causal_kernel<<<S, 256, 0, stream>>>(scores, S, scale);
        // o_h = p @ v_h             [S, VD], K = S ; B[n,k] = v[k, h, n]
        gemm(scores, S, v + h * VD, 1, NH * VD,
             obuf + h * VD, NH * VD, S, VD, S);
    }

    gemm(obuf, NH * VD, w_o, NH * VD, 1, attn, H, T, H, NH * VD);

    // x2 = x1 + attn  (first output)
    add_kernel<<<grid1(T * H), 256, 0, stream>>>(x1, attn, out_x2, T * H);

    // ---------- MoE FFN ----------
    rmsnorm_kernel<<<T, 256, 0, stream>>>(out_x2, nullptr, ln2_w, nullptr, n2, H, H, H);

    gemm(n2, H, gate_w, H, 1, logits, E, T, E, H);
    gate_topk_kernel<<<grid1(T), 256, 0, stream>>>(logits, we, T);

    zero_kernel<<<grid1(T * H), 256, 0, stream>>>(routed, T * H);
    for (int e = 0; e < E; ++e) {
        const float* up = moe_up + (size_t)e * (2 * MI) * H;   // [1024, H]
        const float* dn = moe_dn + (size_t)e * H * MI;         // [H, MI]
        gemm(n2, H, up, H, 1, hup, 2 * MI, T, 2 * MI, H);
        silu_mul_kernel<<<grid1(T * MI), 256, 0, stream>>>(hup, hg, T * MI, MI);
        gemm(hg, MI, dn, MI, 1, hdown, H, T, H, MI);
        scale_add_kernel<<<grid1(T * H), 256, 0, stream>>>(routed, hdown, we, e,
                                                           T * H, H);
    }

    // shared expert
    gemm(n2, H, sh_up, H, 1, hup, 2 * SHI, T, 2 * SHI, H);
    silu_mul_kernel<<<grid1(T * SHI), 256, 0, stream>>>(hup, hg, T * SHI, SHI);
    gemm(hg, SHI, sh_dn, SHI, 1, hdown, H, T, H, SHI);

    // ffn = routed + shared  (second output)
    add_kernel<<<grid1(T * H), 256, 0, stream>>>(routed, hdown, out_ffn, T * H);
}
